// TwoHop_82978768159407
// MI455X (gfx1250) — matrix operand order repaired and checked by reading, not compiled, not run on hardware
//
#include <hip/hip_runtime.h>
#include <hip/hip_bf16.h>
#include <stdint.h>

typedef __attribute__((ext_vector_type(16))) _Float16 v16h;
typedef __attribute__((ext_vector_type(8)))  float    v8f;

#define NN   2048                 // num_nodes (reference constant)
#define NW   64                   // 2048 bits / 32 = words per bitset row
#define EMAX (1u << 20)           // E_MAX
#define FDIM 8

// ---------------- workspace layout (bytes) ----------------
#define OFF_A      ((size_t)0)                         // dense A, f16, N*N
#define SZ_A       ((size_t)NN * NN * 2)
#define OFF_AT     (OFF_A + SZ_A)                      // dense A^T, f16, N*N
#define OFF_SLOT   (OFF_AT + SZ_A)                     // int32 slot map N*N (INT_MAX = empty)
#define SZ_SLOT    ((size_t)NN * NN * 4)
#define OFF_ACCUM  (OFF_SLOT + SZ_SLOT)                // f32 accum, up to 64K edges * F
#define SZ_ACCUM   ((size_t)65536 * FDIM * 4)
#define OFF_COMB   (OFF_ACCUM + SZ_ACCUM)              // combined bitset, N*NW u32
#define SZ_COMB    ((size_t)NN * NW * 4)
#define OFF_ROWCNT (OFF_COMB + SZ_COMB)                // u32[N]
#define OFF_ROWOFF (OFF_ROWCNT + (size_t)NN * 4)       // u32[N+1]
#define WS_WORDS   ((uint32_t)((OFF_ROWOFF + (size_t)(NN + 1) * 4) / 4))

// ---------------- kernels ----------------

__global__ void k_init_ws(uint32_t* ws, uint32_t total, uint32_t slot_lo, uint32_t slot_hi) {
    uint32_t i = blockIdx.x * blockDim.x + threadIdx.x;
    if (i >= total) return;
    ws[i] = (i >= slot_lo && i < slot_hi) ? 0x7FFFFFFFu : 0u;
}

__global__ void k_init_out(uint32_t* out, uint32_t total) {
    uint32_t i = blockIdx.x * blockDim.x + threadIdx.x;
    if (i >= total) return;
    out[i] = (i < 2u * EMAX) ? 0xFFFFFFFFu : 0u;   // -1 pad for edge_index, 0 for attrs
}

__global__ void k_scatter(const int* __restrict__ ei, int E,
                          _Float16* __restrict__ A, _Float16* __restrict__ AT,
                          int* __restrict__ slot, unsigned* __restrict__ comb) {
    int e = blockIdx.x * blockDim.x + threadIdx.x;
    if (e >= E) return;
    int s = ei[e], d = ei[E + e];
    A [(size_t)s * NN + d] = (_Float16)1.0f;
    AT[(size_t)d * NN + s] = (_Float16)1.0f;
    atomicOr(&comb[s * NW + (d >> 5)], 1u << (d & 31));
    atomicMin(&slot[(size_t)s * NN + d], e);           // lowest edge index claims (deterministic)
}

__global__ void k_attr(const int* __restrict__ ei, const float* __restrict__ ea,
                       int E, int F, const int* __restrict__ slot,
                       float* __restrict__ accum) {
    int e = blockIdx.x * blockDim.x + threadIdx.x;
    if (e >= E) return;
    int s = ei[e], d = ei[E + e];
    int c = slot[(size_t)s * NN + d];                  // claimed slot (may be e itself)
    for (int f = 0; f < F; ++f)
        atomicAdd(&accum[(size_t)c * F + f], ea[(size_t)e * F + f]);
}

// one wave32 per 16x16 C tile; K = 2048 in steps of 32 via v_wmma_f32_16x16x32_f16
__global__ void __launch_bounds__(128) k_bmm_struct(const _Float16* __restrict__ A,
                                                    const _Float16* __restrict__ AT,
                                                    unsigned* __restrict__ comb) {
    int wave = threadIdx.x >> 5;
    int lane = threadIdx.x & 31;
    int tile = blockIdx.x * 4 + wave;                  // grid sized exactly: 4096*4 = 128*128 tiles
    int tm = tile >> 7, tn = tile & 127;
    int m0 = tm * 16, n0 = tn * 16;
    int half = lane >> 4;                              // 0: lanes 0-15, 1: lanes 16-31
    int mrow = lane & 15;

    const _Float16* arow = A  + (size_t)(m0 + mrow) * NN;  // A row (M = mrow)
    const _Float16* brow = AT + (size_t)(n0 + mrow) * NN;  // B column (N = mrow) via A^T row

    v8f acc = {};
    for (int kk = 0; kk < NN; kk += 32) {
        union { v16h v; uint32_t u[8]; } a, b;
#pragma unroll
        for (int r = 0; r < 8; ++r) {
            // ISA 16-bit A 16x32 layout: VGPR r holds K pair (r&3)*2 + (r>>2)*16 (+8 for upper lanes)
            int ka = kk + (r & 3) * 2 + (r >> 2) * 16 + half * 8;
            // B 32x16 layout matches A: VGPR r holds K pair (r&3)*2 + (r>>2)*16 (+8 for upper lanes)
            int kb = kk + (r & 3) * 2 + (r >> 2) * 16 + half * 8;
            a.u[r] = *(const uint32_t*)(arow + ka);
            b.u[r] = *(const uint32_t*)(brow + kb);
        }
        acc = __builtin_amdgcn_wmma_f32_16x16x32_f16(
                  false, a.v, false, b.v, (short)0, acc, false, false);
    }

    // C layout: VGPR i, lanes 0-15 -> row m0+i, lanes 16-31 -> row m0+8+i, col = n0 + (lane&15).
    // 16 cols per tile land in one aligned 16-bit half of a 32-bit bitset word.
    unsigned wword = (unsigned)(n0 >> 5);
    unsigned wsh   = (unsigned)(n0 & 16);
#pragma unroll
    for (int i = 0; i < 8; ++i) {
        int row = m0 + half * 8 + i;
        int col = n0 + mrow;
        bool hot = (acc[i] > 0.5f) && (row != col);    // two-hop structure, no self loops
        unsigned bal = (unsigned)__ballot(hot);
        if (lane == 0) {
            unsigned lo = bal & 0xFFFFu;
            if (lo) atomicOr(&comb[(m0 + i) * NW + wword], lo << wsh);
        } else if (lane == 16) {
            unsigned hi = (bal >> 16) & 0xFFFFu;
            if (hi) atomicOr(&comb[(m0 + 8 + i) * NW + wword], hi << wsh);
        }
    }
}

__global__ void k_rowcnt(const unsigned* __restrict__ comb, unsigned* __restrict__ rowcnt) {
    int r = blockIdx.x * blockDim.x + threadIdx.x;
    if (r >= NN) return;
    unsigned c = 0;
    for (int w = 0; w < NW; ++w) c += __popc(comb[r * NW + w]);
    rowcnt[r] = c;
}

__global__ void k_scan(const unsigned* __restrict__ rowcnt, unsigned* __restrict__ rowoff) {
    if (blockIdx.x == 0 && threadIdx.x == 0) {
        unsigned acc = 0;
        for (int r = 0; r < NN; ++r) { rowoff[r] = acc; acc += rowcnt[r]; }
        rowoff[NN] = acc;
    }
}

// one thread per (row, bitset word): emit in row-major, bit-ascending order == jnp.nonzero order
__global__ void k_emit(const unsigned* __restrict__ comb, const unsigned* __restrict__ rowoff,
                       const int* __restrict__ slot, const float* __restrict__ accum,
                       int E, int F,
                       int* __restrict__ out_r, int* __restrict__ out_c,
                       float* __restrict__ out_attr) {
    int tid = blockIdx.x * blockDim.x + threadIdx.x;
    if (tid >= NN * NW) return;
    int r = tid >> 6, w = tid & 63;
    unsigned word = comb[r * NW + w];
    if (!word) return;
    unsigned off = rowoff[r];
    for (int ww = 0; ww < w; ++ww) off += __popc(comb[r * NW + ww]);
    while (word) {
        int b = __ffs(word) - 1;
        word &= word - 1;
        int col = (w << 5) + b;
        if (off < EMAX) {
            out_r[off] = r;
            out_c[off] = col;
            int s = slot[(size_t)r * NN + col];
            float* oa = out_attr + (size_t)off * F;
            if (s < E) {
                for (int f = 0; f < F; ++f) oa[f] = accum[(size_t)s * F + f];
            } else {
                for (int f = 0; f < F; ++f) oa[f] = 0.0f;   // two-hop-only edge: zero attrs
            }
        }
        ++off;
    }
}

// ---------------- launcher ----------------

extern "C" void kernel_launch(void* const* d_in, const int* in_sizes, int n_in,
                              void* d_out, int out_size, void* d_ws, size_t ws_size,
                              hipStream_t stream) {
    const int*   ei = (const int*)d_in[1];     // edge_index [2, E]
    const float* ea = (const float*)d_in[2];   // edge_attr  [E, F]
    int E = in_sizes[1] / 2;
    int F = (E > 0) ? (in_sizes[2] / E) : FDIM;

    char* ws = (char*)d_ws;
    _Float16* A      = (_Float16*)(ws + OFF_A);
    _Float16* AT     = (_Float16*)(ws + OFF_AT);
    int*      slot   = (int*)     (ws + OFF_SLOT);
    float*    accum  = (float*)   (ws + OFF_ACCUM);
    unsigned* comb   = (unsigned*)(ws + OFF_COMB);
    unsigned* rowcnt = (unsigned*)(ws + OFF_ROWCNT);
    unsigned* rowoff = (unsigned*)(ws + OFF_ROWOFF);

    int* out_r = (int*)d_out;                  // [E_MAX]
    int* out_c = out_r + EMAX;                 // [E_MAX]
    float* out_attr = (float*)d_out + 2u * EMAX; // [E_MAX, F]

    const int B = 256;

    // 1) init workspace (slot map -> INT_MAX, everything else -> 0)
    {
        uint32_t total = WS_WORDS;
        uint32_t slot_lo = (uint32_t)(OFF_SLOT / 4);
        uint32_t slot_hi = (uint32_t)((OFF_SLOT + SZ_SLOT) / 4);
        k_init_ws<<<(total + B - 1) / B, B, 0, stream>>>((uint32_t*)d_ws, total, slot_lo, slot_hi);
    }
    // 2) init output padding (-1 for indices, 0 for attrs)
    {
        uint32_t total = (uint32_t)out_size;
        k_init_out<<<(total + B - 1) / B, B, 0, stream>>>((uint32_t*)d_out, total);
    }
    // 3) scatter edges into dense f16 A / A^T, orig bits, slot claims
    k_scatter<<<(E + B - 1) / B, B, 0, stream>>>(ei, E, A, AT, slot, comb);
    // 4) accumulate duplicate edge attrs into claimed slots
    k_attr<<<(E + B - 1) / B, B, 0, stream>>>(ei, ea, E, F, slot, accum);
    // 5) structural A@A via WMMA, OR two-hop bits into combined bitset
    k_bmm_struct<<<(NN / 16) * (NN / 16) / 4, 128, 0, stream>>>(A, AT, comb);
    // 6) row popcounts
    k_rowcnt<<<(NN + B - 1) / B, B, 0, stream>>>(comb, rowcnt);
    // 7) exclusive prefix scan over rows (tiny, serial)
    k_scan<<<1, 1, 0, stream>>>(rowcnt, rowoff);
    // 8) emit edges + gathered attrs in lexicographic order
    k_emit<<<(NN * NW + B - 1) / B, B, 0, stream>>>(comb, rowoff, slot, accum, E, F,
                                                    out_r, out_c, out_attr);
}
